// CrosscoderModule_85040352461497
// MI455X (gfx1250) — matrix operand order repaired and checked by reading, not compiled, not run on hardware
//
#include <hip/hip_runtime.h>

// ---------------------------------------------------------------------------
// Crosscoder SAE forward for MI455X (gfx1250, wave32, WMMA + async-to-LDS).
//   pre  = x(4096x2048) @ W_enc(2048x32768) + b_enc        (bf16 WMMA, f32 acc)
//   topk = top-32 per row, ReLU
//   out  = sum_j val_j * W_dec[idx_j] + b_dec              (gather)
// W_enc (256 MB) is the roofline stream: it is moved HBM->LDS with
// GLOBAL_LOAD_ASYNC_TO_LDS_B128 (ASYNCcnt), double buffered, then converted
// f32->bf16 and swizzled into the CDNA5 WMMA fragment layout inside LDS.
// ---------------------------------------------------------------------------

typedef __attribute__((ext_vector_type(16))) __bf16 v16bf;
typedef __attribute__((ext_vector_type(8)))  float  v8f;
typedef __attribute__((ext_vector_type(4)))  float  f32x4;

#define BATCH   4096
#define KDIM    2048     // N_POS * D_MODEL
#define DSAE    32768
#define DMOD2   2048     // N_POS * D_MODEL (decode row length)
#define TOPK    32

#if defined(__gfx1250__) && __has_builtin(__builtin_amdgcn_global_load_async_to_lds_b128)
#define HAVE_ASYNC_LDS 1
#else
#define HAVE_ASYNC_LDS 0
#endif

#if HAVE_ASYNC_LDS
__device__ __forceinline__ void async_ld_b128(const void* g, void* l) {
    // Prototype (from hipcc diagnostic): param 0/1 are `int __vector(4) *`.
    typedef int v4i __attribute__((vector_size(16)));
    __builtin_amdgcn_global_load_async_to_lds_b128((v4i*)g, (v4i*)l, 0, 0);
}
#if __has_builtin(__builtin_amdgcn_s_wait_asynccnt)
#define WAIT_ASYNC(n) __builtin_amdgcn_s_wait_asynccnt(n)
#else
#define WAIT_ASYNC(n) asm volatile("s_wait_asynccnt %0" :: "n"(n) : "memory")
#endif
#endif

__device__ __forceinline__ unsigned int pack_bf16x2(float a, float b) {
    __bf16 x = (__bf16)a, y = (__bf16)b;
    unsigned short ux, uy;
    __builtin_memcpy(&ux, &x, 2);
    __builtin_memcpy(&uy, &y, 2);
    return ((unsigned int)uy << 16) | (unsigned int)ux;
}

// ---------------------------------------------------------------------------
// Kernel 1: encode GEMM, 128x128 block tile, K-step 32, 8 waves (2M x 4N),
// each wave owns a 64x32 sub-tile = 4x2 WMMA tiles of 16x16.
// LDS fragment layouts (ISA 7.12.2, 16-bit operands):
//   A (16x32): lane = (m&15) + 16*bit3(k); slot = (bit4(k)<<3)|(k&7)
//   B (32x16): lane = (n&15) + 16*bit3(k); slot = (bit4(k)<<3)|(k&7)
// ---------------------------------------------------------------------------
__global__ __launch_bounds__(256) void encode_gemm(
    const float* __restrict__ X,      // (4096, 2048)
    const float* __restrict__ W,      // (2048, 32768)
    const float* __restrict__ benc,   // (32768)
    float*       __restrict__ pre)    // (4096, 32768)
{
    __shared__ float        Braw[2][32][128];  // 32 KB: async W_enc staging (f32)
    __shared__ unsigned int Asw[8][32][8];     //  8 KB: A fragments (2x bf16 / uint)
    __shared__ __bf16       Bsw[8][32][16];    //  8 KB: B fragments

    const int t     = threadIdx.x;
    const int lane  = t & 31;
    const int wave  = t >> 5;
    const int waveM = wave >> 2;             // 0..1
    const int waveN = wave & 3;              // 0..3
    const int row0  = blockIdx.x * 128;      // gridDim.x = 32  (M fastest: L2-share W_enc)
    const int col0  = blockIdx.y * 128;      // gridDim.y = 256 (N)

    v8f acc[4][2];
    const v8f vzero = {0.f, 0.f, 0.f, 0.f, 0.f, 0.f, 0.f, 0.f};
#pragma unroll
    for (int i = 0; i < 4; ++i)
#pragma unroll
        for (int j = 0; j < 2; ++j) acc[i][j] = vzero;

    // Loader mapping: 256 threads move 4096 floats of A and 4096 of B per step.
    const int am = t >> 1;            // A row within chunk, 0..127
    const int ac = (t & 1) * 16;      // A k-offset base (16 consecutive floats)
    const int bk = t >> 3;            // B k-row, 0..31
    const int bn = (t & 7) * 16;      // B col base (16 consecutive floats)

    const float* Abase = X + (size_t)(row0 + am) * KDIM + ac;
    const float* Bbase = W + (size_t)bk * DSAE + col0 + bn;

    const int amtile = am >> 4;
    const int aml    = am & 15;
    const int bntile = bn >> 4;
    const int blaneH = ((bk >> 3) & 1) << 4;
    const int bslot  = (((bk >> 4) & 1) << 3) | (bk & 7);

#if HAVE_ASYNC_LDS
    // ---- prologue: async-stage W_enc chunk 0 (4 b128 per thread per chunk) ----
    {
        const float* g = Bbase;
        float*       l = &Braw[0][bk][bn];
#pragma unroll
        for (int i = 0; i < 4; ++i) async_ld_b128(g + 4 * i, l + 4 * i);
    }
#endif

    for (int i = 0; i < KDIM / 32; ++i) {
        const int k0  = i * 32;
        const int buf = i & 1;

        // ---- A chunk: global -> VGPR (small, L2-resident stream) ----
        float af[16];
        {
            const f32x4* ap = (const f32x4*)(Abase + k0);
#pragma unroll
            for (int j = 0; j < 4; ++j) {
                f32x4 v = ap[j];
                af[4*j+0] = v.x; af[4*j+1] = v.y; af[4*j+2] = v.z; af[4*j+3] = v.w;
            }
        }
        if (k0 + 32 < KDIM) __builtin_prefetch(Abase + k0 + 32, 0, 1);

#if HAVE_ASYNC_LDS
        // ---- issue async W_enc chunk i+1, then wait for chunk i (in-order) ----
        if (k0 + 32 < KDIM) {
            const float* g = Bbase + (size_t)(k0 + 32) * DSAE;
            float*       l = &Braw[buf ^ 1][bk][bn];
#pragma unroll
            for (int j = 0; j < 4; ++j) async_ld_b128(g + 4 * j, l + 4 * j);
            WAIT_ASYNC(4);
        } else {
            WAIT_ASYNC(0);
        }
#endif
        __syncthreads();   // Braw[buf] ready block-wide; prev wmma done reading Asw/Bsw

        // ---- convert + fragment-swizzle A: VGPR -> LDS ----
#pragma unroll
        for (int p = 0; p < 8; ++p) {
            const int kk    = ac + 2 * p;
            const int laneD = aml + (((kk >> 3) & 1) << 4);
            const int sp    = ((kk >> 4) << 2) | ((kk & 7) >> 1);
            Asw[amtile][laneD][sp] = pack_bf16x2(af[2*p], af[2*p+1]);
        }

        // ---- convert + fragment-swizzle B ----
#if HAVE_ASYNC_LDS
#pragma unroll
        for (int j = 0; j < 4; ++j) {
            f32x4 v = *(const f32x4*)&Braw[buf][bk][bn + 4 * j];
            Bsw[bntile][(4*j+0) + blaneH][bslot] = (__bf16)v.x;
            Bsw[bntile][(4*j+1) + blaneH][bslot] = (__bf16)v.y;
            Bsw[bntile][(4*j+2) + blaneH][bslot] = (__bf16)v.z;
            Bsw[bntile][(4*j+3) + blaneH][bslot] = (__bf16)v.w;
        }
#else
        // fallback: synchronous global -> VGPR -> LDS
        {
            const f32x4* bp = (const f32x4*)(Bbase + (size_t)k0 * DSAE);
#pragma unroll
            for (int j = 0; j < 4; ++j) {
                f32x4 v = bp[j];
                Bsw[bntile][(4*j+0) + blaneH][bslot] = (__bf16)v.x;
                Bsw[bntile][(4*j+1) + blaneH][bslot] = (__bf16)v.y;
                Bsw[bntile][(4*j+2) + blaneH][bslot] = (__bf16)v.z;
                Bsw[bntile][(4*j+3) + blaneH][bslot] = (__bf16)v.w;
            }
        }
#endif
        __syncthreads();

        // ---- WMMA: 8 tiles per wave per K-step ----
#pragma unroll
        for (int nt = 0; nt < 2; ++nt) {
            const int ntile = waveN * 2 + nt;
            v16bf bfrag = *(const v16bf*)&Bsw[ntile][lane][0];
#pragma unroll
            for (int mt = 0; mt < 4; ++mt) {
                const int mtile = waveM * 4 + mt;
                v16bf afrag = *(const v16bf*)&Asw[mtile][lane][0];
                acc[mt][nt] = __builtin_amdgcn_wmma_f32_16x16x32_bf16(
                    false, afrag, false, bfrag, (short)0, acc[mt][nt],
                    false, false);
            }
        }
        __syncthreads();   // before next iteration's conversion overwrites Asw/Bsw
    }

    // ---- epilogue: C layout (lane = n + 16*(m_local>=8), vgpr r = m_local&7) ----
    const int halfrow = (lane >> 4) * 8;
    const int cn      = lane & 15;
#pragma unroll
    for (int mt = 0; mt < 4; ++mt) {
        const int mtile = waveM * 4 + mt;
#pragma unroll
        for (int nt = 0; nt < 2; ++nt) {
            const int ntile = waveN * 2 + nt;
            const int col   = col0 + ntile * 16 + cn;
            const float bias = benc[col];
#pragma unroll
            for (int r = 0; r < 8; ++r) {
                const int row = row0 + mtile * 16 + halfrow + r;
                pre[(size_t)row * DSAE + col] = acc[mt][nt][r] + bias;
            }
        }
    }
}

// ---------------------------------------------------------------------------
// Kernel 2: exact top-32 per row. 1024 threads/block, 32 values per thread in
// registers; 32 rounds of argmax (wave32 shuffle reduce + LDS cross-wave).
// ---------------------------------------------------------------------------
__global__ __launch_bounds__(1024) void topk_rows(
    const float* __restrict__ pre,   // (4096, 32768)
    float*       __restrict__ topv,  // (4096, 32)  relu'd values
    int*         __restrict__ topi)  // (4096, 32)
{
    __shared__ float swv[32];
    __shared__ int   swi[32];
    __shared__ int   bi;

    const int t  = threadIdx.x;
    const int b  = blockIdx.x;
    const int ln = t & 31;
    const int wv = t >> 5;

    const float* row = pre + (size_t)b * DSAE;
    float v[32];
#pragma unroll
    for (int j = 0; j < 32; ++j) v[j] = row[t + (j << 10)];  // coalesced, col = j*1024 + t

    for (int r = 0; r < TOPK; ++r) {
        float m = v[0]; int mj = 0;
#pragma unroll
        for (int j = 1; j < 32; ++j) if (v[j] > m) { m = v[j]; mj = j; }
        int gidx = (mj << 10) + t;

        // wave32 reduce
#pragma unroll
        for (int off = 16; off > 0; off >>= 1) {
            float ov = __shfl_xor(m, off, 32);
            int   oi = __shfl_xor(gidx, off, 32);
            if (ov > m || (ov == m && oi < gidx)) { m = ov; gidx = oi; }
        }
        if (ln == 0) { swv[wv] = m; swi[wv] = gidx; }
        __syncthreads();

        if (t < 32) {
            float m2 = swv[t]; int g2 = swi[t];
#pragma unroll
            for (int off = 16; off > 0; off >>= 1) {
                float ov = __shfl_xor(m2, off, 32);
                int   oi = __shfl_xor(g2, off, 32);
                if (ov > m2 || (ov == m2 && oi < g2)) { m2 = ov; g2 = oi; }
            }
            if (t == 0) {
                bi = g2;
                topv[b * TOPK + r] = (m2 > 0.f) ? m2 : 0.f;  // relu
                topi[b * TOPK + r] = g2;
            }
        }
        __syncthreads();
        if ((bi & 1023) == t) v[bi >> 10] = -__builtin_inff();  // evict winner
        __syncthreads();
    }
}

// ---------------------------------------------------------------------------
// Kernel 3: sparse decode. One block per batch row; float4 gathers of the 32
// selected W_dec rows (8 KB each), bias add, coalesced store.
// ---------------------------------------------------------------------------
__global__ __launch_bounds__(512) void decode_scatter(
    const float* __restrict__ topv,
    const int*   __restrict__ topi,
    const float* __restrict__ Wd,     // (32768, 2048)
    const float* __restrict__ bdec,   // (2048)
    float*       __restrict__ out)    // (4096, 2048)
{
    __shared__ float sv[TOPK];
    __shared__ int   si[TOPK];

    const int b = blockIdx.x;
    const int t = threadIdx.x;
    if (t < TOPK) { sv[t] = topv[b * TOPK + t]; si[t] = topi[b * TOPK + t]; }
    __syncthreads();

    const int off = t * 4;                       // 512 threads * 4 = 2048
    f32x4 acc = *(const f32x4*)(bdec + off);
#pragma unroll 4
    for (int j = 0; j < TOPK; ++j) {
        const float v = sv[j];
        if (j + 1 < TOPK) __builtin_prefetch(Wd + (size_t)si[j + 1] * DMOD2 + off, 0, 1);
        const f32x4 w = *(const f32x4*)(Wd + (size_t)si[j] * DMOD2 + off);
        acc.x += v * w.x; acc.y += v * w.y; acc.z += v * w.z; acc.w += v * w.w;
    }
    *(f32x4*)(out + (size_t)b * DMOD2 + off) = acc;
}

// ---------------------------------------------------------------------------
extern "C" void kernel_launch(void* const* d_in, const int* in_sizes, int n_in,
                              void* d_out, int out_size, void* d_ws, size_t ws_size,
                              hipStream_t stream)
{
    (void)in_sizes; (void)n_in; (void)out_size; (void)ws_size;

    const float* x    = (const float*)d_in[0];  // (4096, 2, 1024)
    const float* Wenc = (const float*)d_in[1];  // (2, 1024, 32768)
    const float* Wdec = (const float*)d_in[2];  // (32768, 2, 1024)
    const float* benc = (const float*)d_in[3];  // (32768)
    const float* bdec = (const float*)d_in[4];  // (2, 1024)
    float*       out  = (float*)d_out;          // (4096, 2, 1024)

    float* pre  = (float*)d_ws;                         // 4096*32768 f32 (512 MB)
    float* topv = pre + (size_t)BATCH * DSAE;           // 4096*32 f32
    int*   topi = (int*)(topv + (size_t)BATCH * TOPK);  // 4096*32 i32

    dim3 g1(BATCH / 128, DSAE / 128);   // M fastest: concurrent blocks share W_enc via L2
    encode_gemm<<<g1, 256, 0, stream>>>(x, Wenc, benc, pre);
    topk_rows<<<BATCH, 1024, 0, stream>>>(pre, topv, topi);
    decode_scatter<<<BATCH, 512, 0, stream>>>(topv, topi, Wdec, bdec, out);
}
